// MultiHeadAttention_Diff_55070070670270
// MI455X (gfx1250) — matrix operand order repaired and checked by reading, not compiled, not run on hardware
//
#include <hip/hip_runtime.h>
#include <hip/hip_bf16.h>

// ---------------- problem constants ----------------
static constexpr int EMB   = 1024;
static constexpr int HD    = 64;
static constexpr int TT    = 2048;   // query length
static constexpr int SS    = 2048;   // key length (before bias token)
static constexpr int BBAT  = 4;      // batch
static constexpr int BH    = 64;     // BBAT * HEADS
static constexpr int SP    = 2080;   // S+1 padded to multiple of 32
static constexpr int SAVG  = 2049;   // S+1 (bias_k/bias_v token appended)
static constexpr int NKT   = SP / 16;  // 130 key tiles

typedef __attribute__((ext_vector_type(16))) __bf16       v16bf;
typedef __attribute__((ext_vector_type(8)))  float        v8f;
typedef __attribute__((ext_vector_type(4)))  unsigned int v4u;

union Frag {
  v16bf          v;
  unsigned short s[16];
  v4u            q[2];
};

// Branchless round-to-nearest-even f32 -> bf16 (inputs are finite in this workload).
__device__ __forceinline__ unsigned short f2bf(float x) {
  unsigned int u = __float_as_uint(x);
  u += 0x7fffu + ((u >> 16) & 1u);
  return (unsigned short)(u >> 16);
}

__device__ __forceinline__ v8f wmma_bf16(const Frag& a, const Frag& b, v8f c) {
  return __builtin_amdgcn_wmma_f32_16x16x32_bf16(false, a.v, false, b.v, (short)0, c,
                                                 false, false);
}

// ---------------- small helper kernels ----------------
__global__ void k_zero(float* __restrict__ p, int n) {
  int i = blockIdx.x * blockDim.x + threadIdx.x;
  if (i < n) p[i] = 0.f;
}

// f32 -> bf16 bulk conversion (weights and activations), vectorized 4-wide.
__global__ void k_cvt(const float* __restrict__ src, unsigned short* __restrict__ dst, int n4) {
  int i = blockIdx.x * blockDim.x + threadIdx.x;
  if (i < n4) {
    const float4 f = reinterpret_cast<const float4*>(src)[i];
    ushort4 o;
    o.x = f2bf(f.x); o.y = f2bf(f.y); o.z = f2bf(f.z); o.w = f2bf(f.w);
    reinterpret_cast<ushort4*>(dst)[i] = o;
  }
}

// Fill appended bias_k row / bias_v column and zero the padding tail.
__global__ void k_pad(const float* __restrict__ bk, const float* __restrict__ bv,
                      unsigned short* __restrict__ Kh, unsigned short* __restrict__ Vt) {
  const int bh = blockIdx.x;           // b*16 + h
  const int d  = threadIdx.x;          // 0..63
  const int h  = bh & 15;
  const unsigned short kb = f2bf(bk[h * HD + d]);
  const unsigned short vb = f2bf(bv[h * HD + d]);
  for (int s = SS; s < SP; ++s) {
    Kh[((long)bh * SP + s) * HD + d] = (s == SS) ? kb : (unsigned short)0;
    Vt[((long)bh * HD + d) * SP + s] = (s == SS) ? vb : (unsigned short)0;
  }
}

// ---------------- fused projection / output GEMM (bf16 x bf16 -> f32) ----------------
// C[m,n] = (sum_k A[m,k] * W[n,k] + bias[n]) * scale      (A, W both K-contiguous)
// mode 0: -> Qh  bf16 [bh][t][d]   (scale = 1/sqrt(HD))
// mode 1: -> Kh  bf16 [bh][s][d]
// mode 2: -> Vt  bf16 [bh][d][s]   (transposed for P*V)
// mode 3: -> d_out f32 [m][n]
__global__ __launch_bounds__(256) void k_gemm(
    const unsigned short* __restrict__ Xbf, const unsigned short* __restrict__ Wb,
    const float* __restrict__ bias, float scale, int mode,
    unsigned short* __restrict__ dstb, float* __restrict__ dstf) {
  const int wave = threadIdx.x >> 5;
  const int lane = threadIdx.x & 31;
  const int ph   = lane >> 4;          // lane half
  const int r    = lane & 15;
  const int m0   = blockIdx.y * 128 + wave * 16;
  const int n0   = blockIdx.x * 64;
  const int row  = m0 + r;

  const unsigned short* Arow = Xbf + (long)row * EMB;
  const unsigned short* Wrow[4];
#pragma unroll
  for (int nt = 0; nt < 4; ++nt)
    Wrow[nt] = Wb + (long)(n0 + nt * 16 + r) * EMB + ph * 16;

  auto loadA = [&](Frag& a, int k0) {
    a.q[0] = *reinterpret_cast<const v4u*>(Arow + k0 + ph * 16);
    a.q[1] = *reinterpret_cast<const v4u*>(Arow + k0 + ph * 16 + 8);
  };
  auto loadB = [&](Frag* bf, int k0) {
#pragma unroll
    for (int nt = 0; nt < 4; ++nt) {
      bf[nt].q[0] = *reinterpret_cast<const v4u*>(Wrow[nt] + k0);
      bf[nt].q[1] = *reinterpret_cast<const v4u*>(Wrow[nt] + k0 + 8);
    }
  };

  v8f acc[4] = {};
  Frag a0, a1, b0[4], b1[4];
  loadA(a0, 0);
  loadB(b0, 0);
  // software-pipelined main loop: loads for stage k+32 are in flight during WMMAs for k
  for (int k0 = 0; k0 < EMB - 64; k0 += 64) {
    loadA(a1, k0 + 32); loadB(b1, k0 + 32);
#pragma unroll
    for (int nt = 0; nt < 4; ++nt) acc[nt] = wmma_bf16(a0, b0[nt], acc[nt]);
    loadA(a0, k0 + 64); loadB(b0, k0 + 64);
#pragma unroll
    for (int nt = 0; nt < 4; ++nt) acc[nt] = wmma_bf16(a1, b1[nt], acc[nt]);
  }
  loadA(a1, EMB - 32); loadB(b1, EMB - 32);
#pragma unroll
  for (int nt = 0; nt < 4; ++nt) acc[nt] = wmma_bf16(a0, b0[nt], acc[nt]);
#pragma unroll
  for (int nt = 0; nt < 4; ++nt) acc[nt] = wmma_bf16(a1, b1[nt], acc[nt]);

#pragma unroll
  for (int nt = 0; nt < 4; ++nt) {
    const int col = n0 + nt * 16 + r;
    const float bval = bias[col];
#pragma unroll
    for (int v = 0; v < 8; ++v) {
      const int m = m0 + 8 * ph + v;                 // C layout: rows 0-7 / 8-15 per half
      const float val = (acc[nt][v] + bval) * scale;
      if (mode == 3) {
        dstf[(long)m * EMB + col] = val;
      } else {
        const int t = m >> 2, b = m & 3;             // row = t*B + b
        const int h = col >> 6, d = col & 63;
        const int bh = b * 16 + h;
        if (mode == 0)      dstb[((long)bh * TT + t) * HD + d] = f2bf(val);
        else if (mode == 1) dstb[((long)bh * SP + t) * HD + d] = f2bf(val);
        else                dstb[((long)bh * HD + d) * SP + t] = f2bf(val);
      }
    }
  }
}

// ---------------- attention core (two-pass softmax + WMMA) ----------------
__global__ __launch_bounds__(128) void k_attn(
    const unsigned short* __restrict__ Qh, const unsigned short* __restrict__ Kh,
    const unsigned short* __restrict__ Vt, unsigned short* __restrict__ AO,
    float* __restrict__ avg) {
  __shared__ __align__(16) unsigned short Plds[4][2][16 * 32];  // per-wave double buffer

  const int bh   = blockIdx.x;
  const int b    = bh >> 4, h = bh & 15;
  const int wave = threadIdx.x >> 5;
  const int lane = threadIdx.x & 31;
  const int ph   = lane >> 4;
  const int r    = lane & 15;
  const int q0   = blockIdx.y * 64 + wave * 16;

  // Q fragments for this wave's 16 rows (K-dim 64 -> two x32 fragments), reused everywhere.
  const unsigned short* qrow = Qh + ((long)bh * TT + q0 + r) * HD;
  Frag aq0, aq1;
  aq0.q[0] = *reinterpret_cast<const v4u*>(qrow + ph * 16);
  aq0.q[1] = *reinterpret_cast<const v4u*>(qrow + ph * 16 + 8);
  aq1.q[0] = *reinterpret_cast<const v4u*>(qrow + 32 + ph * 16);
  aq1.q[1] = *reinterpret_cast<const v4u*>(qrow + 40 + ph * 16);

  // base for this lane's K columns (column r, hd-slice starting at ph*16)
  const unsigned short* Kbase = Kh + ((long)bh * SP + r) * HD + ph * 16;
  auto loadK = [&](Frag* f, int kt) {    // f[0]: d 0..31, f[1]: d 32..63
    const unsigned short* kc = Kbase + (long)kt * 16 * HD;
    f[0].q[0] = *reinterpret_cast<const v4u*>(kc);
    f[0].q[1] = *reinterpret_cast<const v4u*>(kc + 8);
    f[1].q[0] = *reinterpret_cast<const v4u*>(kc + 32);
    f[1].q[1] = *reinterpret_cast<const v4u*>(kc + 40);
  };

  float mrow[8], lrow[8];
#pragma unroll
  for (int v = 0; v < 8; ++v) { mrow[v] = -1e30f; lrow[v] = 0.f; }

  // update running softmax stats from one 16x16 score tile
  auto stat = [&](Frag* f, int kt) {
    v8f c = {};
    c = wmma_bf16(aq0, f[0], c);
    c = wmma_bf16(aq1, f[1], c);
    const bool valid = (kt * 16 + r) < SAVG;
#pragma unroll
    for (int v = 0; v < 8; ++v) {
      float x = valid ? c[v] : -1e30f;
      float xm = x;
      xm = fmaxf(xm, __shfl_xor(xm, 1, 32));
      xm = fmaxf(xm, __shfl_xor(xm, 2, 32));
      xm = fmaxf(xm, __shfl_xor(xm, 4, 32));
      xm = fmaxf(xm, __shfl_xor(xm, 8, 32));
      const float nm = fmaxf(mrow[v], xm);
      float e = valid ? __expf(x - nm) : 0.f;
      e += __shfl_xor(e, 1, 32);
      e += __shfl_xor(e, 2, 32);
      e += __shfl_xor(e, 4, 32);
      e += __shfl_xor(e, 8, 32);
      lrow[v] = lrow[v] * __expf(mrow[v] - nm) + e;
      mrow[v] = nm;
    }
  };

  // ---- pass 1: running row-max and softmax denominator (pipelined by tile pairs) ----
  Frag kA[2], kB[2];
  loadK(kA, 0);
  for (int kt = 0; kt < NKT - 2; kt += 2) {
    loadK(kB, kt + 1);
    stat(kA, kt);
    loadK(kA, kt + 2);
    stat(kB, kt + 1);
  }
  loadK(kB, NKT - 1);
  stat(kA, NKT - 2);
  stat(kB, NKT - 1);

  float linv[8];
#pragma unroll
  for (int v = 0; v < 8; ++v) linv[v] = 1.f / lrow[v];

  // ---- pass 2: probabilities -> avg_weights column sums + P x V ----
  v8f o[4] = {};
  for (int kb32 = 0; kb32 < SP / 32; ++kb32) {
    const int kbase = kb32 * 32;
    // issue ALL global loads of this iteration first (2 K sub-tiles + 4 V fragments)
    Frag kf0[2], kf1[2], vf[4];
    loadK(kf0, kb32 * 2);
    loadK(kf1, kb32 * 2 + 1);
#pragma unroll
    for (int nt = 0; nt < 4; ++nt) {
      const unsigned short* vr = Vt + ((long)(bh * HD + nt * 16 + r)) * SP + kbase + ph * 16;
      vf[nt].q[0] = *reinterpret_cast<const v4u*>(vr);
      vf[nt].q[1] = *reinterpret_cast<const v4u*>(vr + 8);
    }
    unsigned short* pl = &Plds[wave][kb32 & 1][0];
    // probabilities for the two 16-column sub-tiles
#pragma unroll
    for (int sub = 0; sub < 2; ++sub) {
      const int kt = kb32 * 2 + sub;
      v8f c = {};
      c = wmma_bf16(aq0, (sub ? kf1 : kf0)[0], c);
      c = wmma_bf16(aq1, (sub ? kf1 : kf0)[1], c);
      const int kcol = kt * 16 + r;
      const bool valid = kcol < SAVG;
      float csum = 0.f;
#pragma unroll
      for (int v = 0; v < 8; ++v) {
        const float p = valid ? __expf(c[v] - mrow[v]) * linv[v] : 0.f;
        csum += p;
        pl[(8 * ph + v) * 32 + sub * 16 + r] = f2bf(p);
      }
      csum += __shfl_xor(csum, 16, 32);
      if (ph == 0 && valid)
        atomicAdd(&avg[(long)bh * SAVG + kcol], csum * 0.0625f);  // /HEADS
    }
    __syncthreads();  // make cross-lane P tile visible / order LDS store->load
    Frag ap;
    const unsigned short* pr = pl + r * 32;
    ap.q[0] = *reinterpret_cast<const v4u*>(pr + ph * 16);
    ap.q[1] = *reinterpret_cast<const v4u*>(pr + ph * 16 + 8);
#pragma unroll
    for (int nt = 0; nt < 4; ++nt) o[nt] = wmma_bf16(ap, vf[nt], o[nt]);
  }

  // store attention output (bf16) for the final projection GEMM
#pragma unroll
  for (int nt = 0; nt < 4; ++nt) {
    const int col = h * HD + nt * 16 + r;
#pragma unroll
    for (int v = 0; v < 8; ++v) {
      const int t = q0 + 8 * ph + v;
      AO[((long)(t * BBAT + b)) * EMB + col] = f2bf(o[nt][v]);
    }
  }
}

// ---------------- host launch ----------------
extern "C" void kernel_launch(void* const* d_in, const int* in_sizes, int n_in,
                              void* d_out, int out_size, void* d_ws, size_t ws_size,
                              hipStream_t stream) {
  (void)in_sizes; (void)n_in; (void)out_size; (void)ws_size;
  const float* query = (const float*)d_in[0];
  const float* key   = (const float*)d_in[1];
  const float* value = (const float*)d_in[2];
  const float* qpw   = (const float*)d_in[3];
  const float* kpw   = (const float*)d_in[4];
  const float* vpw   = (const float*)d_in[5];
  const float* ipb   = (const float*)d_in[6];
  const float* bk    = (const float*)d_in[7];
  const float* bv    = (const float*)d_in[8];
  const float* opw   = (const float*)d_in[9];
  const float* opb   = (const float*)d_in[10];

  // workspace layout (bf16 / ushort units), ~106 MB total.
  // AO (attention output) aliases Xq, which is dead after the Q projection.
  constexpr size_t NW  = (size_t)EMB * EMB;        // 1M  (weights)
  constexpr size_t NX  = (size_t)TT * BBAT * EMB;  // 8M  (activations)
  unsigned short* ws = (unsigned short*)d_ws;
  unsigned short* Wq = ws;
  unsigned short* Wk = Wq + NW;
  unsigned short* Wv = Wk + NW;
  unsigned short* Wo = Wv + NW;
  unsigned short* Xq = Wo + NW;
  unsigned short* Xk = Xq + NX;
  unsigned short* Xv = Xk + NX;
  unsigned short* Qh = Xv + NX;
  unsigned short* Kh = Qh + (size_t)BH * TT * HD;
  unsigned short* Vt = Kh + (size_t)BH * SP * HD;
  unsigned short* AO = Xq;                          // alias (Xq dead after Q proj)

  float* outp = (float*)d_out;
  float* avg  = outp + (size_t)TT * BBAT * EMB;

  k_zero<<<(BH * SAVG + 255) / 256, 256, 0, stream>>>(avg, BH * SAVG);
  // bf16 conversions: 4 weight matrices + 3 activation tensors (all sizes /4 for float4)
  k_cvt<<<(int)(NW / 4 / 256), 256, 0, stream>>>(qpw, Wq, (int)(NW / 4));
  k_cvt<<<(int)(NW / 4 / 256), 256, 0, stream>>>(kpw, Wk, (int)(NW / 4));
  k_cvt<<<(int)(NW / 4 / 256), 256, 0, stream>>>(vpw, Wv, (int)(NW / 4));
  k_cvt<<<(int)(NW / 4 / 256), 256, 0, stream>>>(opw, Wo, (int)(NW / 4));
  k_cvt<<<(int)(NX / 4 / 256), 256, 0, stream>>>(query, Xq, (int)(NX / 4));
  k_cvt<<<(int)(NX / 4 / 256), 256, 0, stream>>>(key,   Xk, (int)(NX / 4));
  k_cvt<<<(int)(NX / 4 / 256), 256, 0, stream>>>(value, Xv, (int)(NX / 4));
  k_pad<<<BH, HD, 0, stream>>>(bk, bv, Kh, Vt);

  const float scaling = 0.125f;  // HD^-0.5
  k_gemm<<<dim3(EMB / 64, (TT * BBAT) / 128), 256, 0, stream>>>(
      Xq, Wq, ipb, scaling, 0, Qh, nullptr);
  k_gemm<<<dim3(EMB / 64, (SS * BBAT) / 128), 256, 0, stream>>>(
      Xk, Wk, ipb + EMB, 1.f, 1, Kh, nullptr);
  k_gemm<<<dim3(EMB / 64, (SS * BBAT) / 128), 256, 0, stream>>>(
      Xv, Wv, ipb + 2 * EMB, 1.f, 2, Vt, nullptr);

  k_attn<<<dim3(BH, TT / 64), 128, 0, stream>>>(Qh, Kh, Vt, AO, avg);

  k_gemm<<<dim3(EMB / 64, (TT * BBAT) / 128), 256, 0, stream>>>(
      AO, Wo, opb, 1.f, 3, nullptr, outp);
}
